// GroupedRecurrentMultiHeadAttention_78941498900918
// MI455X (gfx1250) — matrix operand order repaired and checked by reading, not compiled, not run on hardware
//
#include <hip/hip_runtime.h>
#include <hip/hip_bf16.h>
#include <math.h>

// ---------------------------------------------------------------------------
// GroupedRecurrentMultiHeadAttention for MI455X (gfx1250, wave32, WMMA)
//   D_MODEL=1024, N_QUERY=4, N_HEAD=8, D_HEAD=32, D_KV=256, B=4, S=1024
// All matmuls: v_wmma_f32_16x16x32_bf16 (fp32 accumulate).
// Projection GEMM: 128x64 tile, async-DMA (ASYNCcnt) double-buffered LDS,
// branch-free unroll-2 pipeline. V is stored transposed so the attention
// t-loop needs no workgroup barriers at all.
// ---------------------------------------------------------------------------

typedef __bf16 bf16_t;
typedef __attribute__((ext_vector_type(16))) __bf16 v16bf;
typedef __attribute__((ext_vector_type(8)))  __bf16 v8bf;
typedef __attribute__((ext_vector_type(8)))  float  v8f;

#define NB   4
#define SEQ  1024
#define DM   1024
#define NQ   4
#define NH   8
#define DH   32
#define DKV  256
#define MTOK (NB * SEQ)          // 4096 token rows

static __device__ __forceinline__ v16bf frag_cat(v8bf lo, v8bf hi) {
    v16bf r;
#pragma unroll
    for (int i = 0; i < 8; ++i) { r[i] = lo[i]; r[i + 8] = hi[i]; }
    return r;
}

// A fragment (16x32 bf16, MxK): lane holds row m = m0 + lane%16,
// K chunks at k0 + (lane/16)*8 and +16   (ISA 7.12.2 16-bit A table)
static __device__ __forceinline__ v16bf load_a_frag(const bf16_t* A, int ld,
                                                    int m0, int k0, int lane) {
    const bf16_t* p = A + (m0 + (lane & 15)) * ld + k0 + ((lane >> 4) << 3);
    v8bf lo = *(const v8bf*)p;
    v8bf hi = *(const v8bf*)(p + 16);
    return frag_cat(lo, hi);
}

// B fragment (32x16 bf16, KxN) from an N-major (transposed) matrix:
// lane holds column n = n0 + lane%16, K chunks at k0 + (lane/16)*8 and +16
static __device__ __forceinline__ v16bf load_b_frag(const bf16_t* Bt, int ld,
                                                    int n0, int k0, int lane) {
    const bf16_t* p = Bt + (n0 + (lane & 15)) * ld + k0 + ((lane >> 4) << 3);
    v8bf lo = *(const v8bf*)p;
    v8bf hi = *(const v8bf*)(p + 16);
    return frag_cat(lo, hi);
}

static __device__ __forceinline__ v8f wmma_bf16(v16bf a, v16bf b, v8f c) {
    return __builtin_amdgcn_wmma_f32_16x16x32_bf16(false, a, false, b,
                                                   (short)0, c, false, false);
}

static __device__ __forceinline__ void lds_fence() {
#if defined(__HIP_DEVICE_COMPILE__)
    asm volatile("s_wait_dscnt 0" ::: "memory");
#endif
}

// Async DMA: 16 bytes global -> LDS per lane (VGLOBAL encoding, ASYNCcnt)
static __device__ __forceinline__ void async_cp_b128(unsigned lds_off,
                                                     const bf16_t* gptr) {
#if defined(__HIP_DEVICE_COMPILE__)
    asm volatile("global_load_async_to_lds_b128 %0, %1, off"
                 :: "v"(lds_off), "v"(gptr) : "memory");
#endif
}

static __device__ __forceinline__ void wait_async0() {
#if defined(__HIP_DEVICE_COMPILE__)
    asm volatile("s_wait_asynccnt 0" ::: "memory");
#endif
}

// ------------------------------- conversions -------------------------------

__global__ void cvt_bf16(const float* __restrict__ in, bf16_t* __restrict__ out, int n) {
    int i = blockIdx.x * blockDim.x + threadIdx.x;
    if (i < n) out[i] = (bf16_t)in[i];
}

// in: KxN row-major fp32  ->  out: NxK row-major bf16 (transposed)
__global__ void cvt_t_bf16(const float* __restrict__ in, bf16_t* __restrict__ out,
                           int K, int N) {
    int i = blockIdx.x * blockDim.x + threadIdx.x;
    if (i < K * N) {
        int n = i / K;
        int k = i - n * K;
        out[i] = (bf16_t)in[k * N + n];
    }
}

// ------------------------- projection GEMM (bf16) --------------------------
// C[M,N] = A[M,K] x Bt[N,K]^T ; 128x64 block tile, 8 waves (4x2),
// each wave 32x32 (4 WMMAs/k-step). Async-DMA double-buffered LDS,
// compile-time buffer indices, peeled epilogue (K is a multiple of 64).
// out_mode 0: row-major C[M,N] bf16
// out_mode 1: V-transpose VT[b][h][d][s] bf16 (requires N==256)
__global__ void __launch_bounds__(256)
gemm_bf16(const bf16_t* __restrict__ A, const bf16_t* __restrict__ Bt,
          bf16_t* __restrict__ C, int M, int N, int K, int out_mode) {
    __shared__ __align__(16) bf16_t Asm[2][128][40];  // 128 rows x 32 k, pad 40
    __shared__ __align__(16) bf16_t Bsm[2][64][40];

    const int tid  = threadIdx.x;
    const int lane = tid & 31;
    const int wave = tid >> 5;
    const int m_blk = blockIdx.x * 128;
    const int n_blk = blockIdx.y * 64;
    const int wm = (wave >> 1) * 32;       // wave's rows within the tile
    const int wn = (wave & 1) * 32;        // wave's cols within the tile

    // copy mapping: thread -> 2 A rows + 1 B row, one 16B chunk each
    const int crow = tid >> 2;             // 0..63
    const int cchk = (tid & 3) * 8;        // element offset of 16B chunk
    const bf16_t* gA0 = A + (m_blk + crow) * K + cchk;
    const bf16_t* gA1 = A + (m_blk + 64 + crow) * K + cchk;
    const bf16_t* gB  = Bt + (n_blk + crow) * K + cchk;
    unsigned la0[2], la1[2], lb[2];
#pragma unroll
    for (int b = 0; b < 2; ++b) {
        la0[b] = (unsigned)(uintptr_t)&Asm[b][crow][cchk];
        la1[b] = (unsigned)(uintptr_t)&Asm[b][64 + crow][cchk];
        lb[b]  = (unsigned)(uintptr_t)&Bsm[b][crow][cchk];
    }

    v8f acc00 = {}, acc01 = {}, acc10 = {}, acc11 = {};

    auto dma_tile = [&](int buf, int k) {
        async_cp_b128(la0[buf], gA0 + k);
        async_cp_b128(la1[buf], gA1 + k);
        async_cp_b128(lb[buf],  gB + k);
    };
    auto compute_tile = [&](int buf) {
        v16bf a0 = load_a_frag(&Asm[buf][0][0], 40, wm,      0, lane);
        v16bf a1 = load_a_frag(&Asm[buf][0][0], 40, wm + 16, 0, lane);
        v16bf b0 = load_b_frag(&Bsm[buf][0][0], 40, wn,      0, lane);
        v16bf b1 = load_b_frag(&Bsm[buf][0][0], 40, wn + 16, 0, lane);
        acc00 = wmma_bf16(a0, b0, acc00);
        acc01 = wmma_bf16(a0, b1, acc01);
        acc10 = wmma_bf16(a1, b0, acc10);
        acc11 = wmma_bf16(a1, b1, acc11);
    };

    // prologue
    dma_tile(0, 0);
    wait_async0();
    __syncthreads();

    // branch-free unroll-2 steady state: consume one buffer, DMA the other
    for (int k0 = 0; k0 + 64 < K; k0 += 64) {
        dma_tile(1, k0 + 32);
        compute_tile(0);
        wait_async0();
        __syncthreads();
        dma_tile(0, k0 + 64);
        compute_tile(1);
        wait_async0();
        __syncthreads();
    }
    // epilogue: last two k-tiles (K-64, K-32)
    dma_tile(1, K - 32);
    compute_tile(0);
    wait_async0();
    __syncthreads();
    compute_tile(1);

    const int row_off = (lane >> 4) << 3;  // +8 rows for upper half-wave
    const int col     = lane & 15;
#pragma unroll
    for (int r = 0; r < 8; ++r) {
        int row0 = m_blk + wm + r + row_off;
        int row1 = row0 + 16;
        int c0 = n_blk + wn + col;
        int c1 = c0 + 16;
        if (out_mode == 0) {
            C[row0 * N + c0] = (bf16_t)acc00[r];
            C[row0 * N + c1] = (bf16_t)acc01[r];
            C[row1 * N + c0] = (bf16_t)acc10[r];
            C[row1 * N + c1] = (bf16_t)acc11[r];
        } else {
            // VT[((b*NH + h)*DH + d)*SEQ + s], row = b*SEQ + s, col = h*DH + d
            int b0i = row0 >> 10, s0 = row0 & (SEQ - 1);
            int b1i = row1 >> 10, s1 = row1 & (SEQ - 1);
            C[((b0i * NH + (c0 >> 5)) * DH + (c0 & 31)) * SEQ + s0] = (bf16_t)acc00[r];
            C[((b0i * NH + (c1 >> 5)) * DH + (c1 & 31)) * SEQ + s0] = (bf16_t)acc01[r];
            C[((b1i * NH + (c0 >> 5)) * DH + (c0 & 31)) * SEQ + s1] = (bf16_t)acc10[r];
            C[((b1i * NH + (c1 >> 5)) * DH + (c1 & 31)) * SEQ + s1] = (bf16_t)acc11[r];
        }
    }
}

// --------------------- flash attention (per b,q,h group) -------------------
// block = 128 threads (4 waves), each wave owns 16 query rows, fully
// wave-independent (no workgroup barriers): V comes pre-transposed.
__global__ void __launch_bounds__(128)
attn_kernel(const bf16_t* __restrict__ Q, const bf16_t* __restrict__ Kb,
            const bf16_t* __restrict__ VT, float* __restrict__ attn) {
    const int tid  = threadIdx.x;
    const int lane = tid & 31;
    const int wave = tid >> 5;
    const int mt = blockIdx.x;             // 16 tiles of 64 query rows
    const int qg = blockIdx.y >> 3;        // query group
    const int h  = blockIdx.y & 7;         // kv head
    const int b  = blockIdx.z;

    __shared__ __align__(16) bf16_t Pbuf[4][16][40];   // per-wave P tile

    const int m0 = mt * 64 + wave * 16;    // first query row for this wave
    const float scale = 0.17677669529663689f;          // 1/sqrt(32)

    // Q A-fragment: row stride DM, cols qg*256 + h*32 + d
    const bf16_t* qbase = Q + (b * SEQ) * DM + qg * DKV + h * DH;
    v16bf a_q = load_a_frag(qbase, DM, m0, 0, lane);

    const bf16_t* vtbase = VT + (b * NH + h) * DH * SEQ;   // [d][t], ld=SEQ

    v8f o0 = {}, o1 = {};
    float mi[8], li[8];
#pragma unroll
    for (int r = 0; r < 8; ++r) { mi[r] = -INFINITY; li[r] = 0.0f; }

    const int half = lane >> 4;
    const int nl   = lane & 15;

    for (int t0 = 0; t0 < SEQ; t0 += 32) {
        // prefetch next K/V tiles into cache while this tile is consumed
        if (t0 + 32 < SEQ) {
            __builtin_prefetch(Kb + ((b * SEQ + t0 + 32 + lane) * NH + h) * DH, 0, 0);
            __builtin_prefetch(vtbase + lane * SEQ + t0 + 32, 0, 0);
        }

        // ---- scores: S = Q x K^T (two 16-key tiles) ----
        const bf16_t* kbase = Kb + ((b * SEQ + t0) * NH + h) * DH;
        v16bf b_k0 = load_b_frag(kbase, NH * DH, 0,  0, lane);
        v16bf b_k1 = load_b_frag(kbase, NH * DH, 16, 0, lane);
        v8f z = {};
        v8f s0 = wmma_bf16(a_q, b_k0, z);
        v8f s1 = wmma_bf16(a_q, b_k1, z);

        // ---- online softmax per query row (TRANS ops co-exec with WMMA) ----
#pragma unroll
        for (int r = 0; r < 8; ++r) {
            float v0 = s0[r] * scale, v1 = s1[r] * scale;
            float mx = fmaxf(v0, v1);
            mx = fmaxf(mx, __shfl_xor(mx, 8, 16));
            mx = fmaxf(mx, __shfl_xor(mx, 4, 16));
            mx = fmaxf(mx, __shfl_xor(mx, 2, 16));
            mx = fmaxf(mx, __shfl_xor(mx, 1, 16));
            float mnew  = fmaxf(mi[r], mx);
            float alpha = __expf(mi[r] - mnew);
            float p0 = __expf(v0 - mnew);
            float p1 = __expf(v1 - mnew);
            float sum = p0 + p1;
            sum += __shfl_xor(sum, 8, 16);
            sum += __shfl_xor(sum, 4, 16);
            sum += __shfl_xor(sum, 2, 16);
            sum += __shfl_xor(sum, 1, 16);
            li[r] = li[r] * alpha + sum;
            mi[r] = mnew;
            o0[r] *= alpha;
            o1[r] *= alpha;
            Pbuf[wave][r + 8 * half][nl]      = (bf16_t)p0;
            Pbuf[wave][r + 8 * half][16 + nl] = (bf16_t)p1;
        }
        lds_fence();   // P visible to all lanes of this wave

        // ---- O += P x V (B-frags straight from transposed V in global) ----
        v16bf a_p  = load_a_frag(&Pbuf[wave][0][0], 40, 0, 0, lane);
        v16bf b_v0 = load_b_frag(vtbase, SEQ, 0,  t0, lane);
        v16bf b_v1 = load_b_frag(vtbase, SEQ, 16, t0, lane);
        o0 = wmma_bf16(a_p, b_v0, o0);
        o1 = wmma_bf16(a_p, b_v1, o1);
    }

    // ---- normalize and store (layout matches final output) ----
#pragma unroll
    for (int r = 0; r < 8; ++r) {
        float inv = 1.0f / li[r];
        int s = m0 + r + 8 * half;
        int base = (b * SEQ + s) * DM + qg * DKV + h * DH;
        attn[base + nl]      = o0[r] * inv;
        attn[base + 16 + nl] = o1[r] * inv;
    }
}

// ----------------- ELU linear-memory branch + sigmoid gate -----------------
// M = B*S*NQ = 16384 rows of sigma_q (K=256); block = 16 rows, 4 waves x 4 tiles
__global__ void __launch_bounds__(128)
memgate_kernel(const bf16_t* __restrict__ Q, const bf16_t* __restrict__ memT,
               const float* __restrict__ mem_norm, const float* __restrict__ mem_w,
               const float* __restrict__ attn, float* __restrict__ out) {
    const int tid  = threadIdx.x;
    const int lane = tid & 31;
    const int wave = tid >> 5;
    const int m0   = blockIdx.x * 16;      // global sigma row tile

    __shared__ __align__(16) bf16_t sigma[16][264];    // padded K=256 rows
    __shared__ float norms[16];

    // stage sigma = elu(q) for 16 rows; 8 threads per row, 32 cols each
    {
        int row = tid >> 3;
        int c0  = (tid & 7) * 32;
        int m   = m0 + row;
        int qg  = m & 3;
        int bs  = m >> 2;
        const bf16_t* qrow = Q + bs * DM + qg * DKV;
        float part = 0.0f;
#pragma unroll 8
        for (int j = 0; j < 32; ++j) {
            float v = (float)qrow[c0 + j];
            float e = (v > 0.0f) ? v : (__expf(v) - 1.0f);
            sigma[row][c0 + j] = (bf16_t)e;
            part += e * mem_norm[c0 + j];
        }
        part += __shfl_xor(part, 4, 8);
        part += __shfl_xor(part, 2, 8);
        part += __shfl_xor(part, 1, 8);
        if ((tid & 7) == 0) norms[row] = part;
    }
    __syncthreads();

    // 4 WMMA n-tiles per wave over K=256
    v8f acc[4] = {{}, {}, {}, {}};
    for (int k0 = 0; k0 < DKV; k0 += 32) {
        v16bf a = load_a_frag(&sigma[0][0], 264, 0, k0, lane);
#pragma unroll
        for (int i = 0; i < 4; ++i) {
            v16bf bfr = load_b_frag(memT, DKV, wave * 64 + i * 16, k0, lane);
            acc[i] = wmma_bf16(a, bfr, acc[i]);
        }
    }

    const float w  = 1.0f / (1.0f + __expf(-mem_w[0]));
    const float cw = 1.0f - w;
    const int half = lane >> 4;
    const int nl   = lane & 15;
#pragma unroll
    for (int r = 0; r < 8; ++r) {
        int rowl = r + 8 * half;
        int m    = m0 + rowl;
        int qg   = m & 3;
        int bs   = m >> 2;
        float inv = 1.0f / norms[rowl];
#pragma unroll
        for (int i = 0; i < 4; ++i) {
            int n   = wave * 64 + i * 16 + nl;
            int off = bs * DM + qg * DKV + n;
            out[off] = attn[off] * cw + acc[i][r] * inv * w;
        }
    }
}

// ------------------------------- launcher ----------------------------------

extern "C" void kernel_launch(void* const* d_in, const int* in_sizes, int n_in,
                              void* d_out, int out_size, void* d_ws, size_t ws_size,
                              hipStream_t stream) {
    const float* x    = (const float*)d_in[0];
    const float* Wq   = (const float*)d_in[1];
    const float* Wk   = (const float*)d_in[2];
    const float* Wv   = (const float*)d_in[3];
    const float* mem  = (const float*)d_in[4];
    const float* mnrm = (const float*)d_in[5];
    const float* mw   = (const float*)d_in[6];
    float* out = (float*)d_out;

    char* ws = (char*)d_ws;
    size_t off = 0;
    auto alloc = [&](size_t bytes) { char* p = ws + off; off += (bytes + 255) & ~size_t(255); return p; };
    bf16_t* Xbf  = (bf16_t*)alloc((size_t)MTOK * DM * 2);     // 8 MiB
    bf16_t* Wqt  = (bf16_t*)alloc((size_t)DM * DM * 2);       // 2 MiB
    bf16_t* Wkt  = (bf16_t*)alloc((size_t)DKV * DM * 2);      // 0.5 MiB
    bf16_t* Wvt  = (bf16_t*)alloc((size_t)DKV * DM * 2);      // 0.5 MiB
    bf16_t* MemT = (bf16_t*)alloc((size_t)DKV * DKV * 2);     // 128 KiB
    bf16_t* Qbf  = (bf16_t*)alloc((size_t)MTOK * DM * 2);     // 8 MiB
    bf16_t* Kbf  = (bf16_t*)alloc((size_t)MTOK * DKV * 2);    // 2 MiB
    bf16_t* VTbf = (bf16_t*)alloc((size_t)MTOK * DKV * 2);    // 2 MiB (transposed)
    float*  Attn = (float*)alloc((size_t)MTOK * DM * 4);      // 16 MiB
    (void)ws_size; (void)in_sizes; (void)n_in; (void)out_size;

    // 1) fp32 -> bf16 conversions (weights transposed for B-fragment loads)
    cvt_bf16<<<(MTOK * DM + 255) / 256, 256, 0, stream>>>(x, Xbf, MTOK * DM);
    cvt_t_bf16<<<(DM * DM + 255) / 256, 256, 0, stream>>>(Wq, Wqt, DM, DM);
    cvt_t_bf16<<<(DM * DKV + 255) / 256, 256, 0, stream>>>(Wk, Wkt, DM, DKV);
    cvt_t_bf16<<<(DM * DKV + 255) / 256, 256, 0, stream>>>(Wv, Wvt, DM, DKV);
    cvt_t_bf16<<<(DKV * DKV + 255) / 256, 256, 0, stream>>>(mem, MemT, DKV, DKV);

    // 2) projections on WMMA (async-DMA LDS staging); V stored transposed
    gemm_bf16<<<dim3(MTOK / 128, DM / 64), 256, 0, stream>>>(Xbf, Wqt, Qbf, MTOK, DM, DM, 0);
    gemm_bf16<<<dim3(MTOK / 128, DKV / 64), 256, 0, stream>>>(Xbf, Wkt, Kbf, MTOK, DKV, DM, 0);
    gemm_bf16<<<dim3(MTOK / 128, DKV / 64), 256, 0, stream>>>(Xbf, Wvt, VTbf, MTOK, DKV, DM, 1);

    // 3) flash attention per (b, query-group, head)
    attn_kernel<<<dim3(SEQ / 64, NQ * NH, NB), 128, 0, stream>>>(Qbf, Kbf, VTbf, Attn);

    // 4) ELU memory branch + sigmoid gate -> out
    memgate_kernel<<<(MTOK * NQ) / 16, 128, 0, stream>>>(Qbf, MemT, mnrm, mw, Attn, out);
}
